// ThreeWayMVComposition_21260088115478
// MI455X (gfx1250) — matrix operand order repaired and checked by reading, not compiled, not run on hardware
//
#include <hip/hip_runtime.h>
#include <stdint.h>

#define DD    32
#define NPB   8192                 // N per batch
#define BB    4
#define MROWS (BB * NPB)           // 32768 rows
#define KSEG  1024                 // d*d (one shift segment of K)
#define NCOLS 1024                 // output cols of h_mat
#define HVEC_ELEMS (MROWS * DD)    // h_vec floats (comes first in d_out)
#define KSTEPS 96                  // 3072 / 32
#define LDS_LINE 80                // 64B payload + 16B TDM pad per B line
#define BUFBYTES (128 * LDS_LINE)  // one B tile: 128 cols x 80B

typedef __attribute__((ext_vector_type(16))) __bf16 v16bf;
typedef __attribute__((ext_vector_type(8)))  float  v8f;
typedef __attribute__((ext_vector_type(4)))  float  f32x4;
typedef __attribute__((ext_vector_type(4)))  unsigned int v4u;
typedef __attribute__((ext_vector_type(8)))  int v8i_;
typedef __attribute__((ext_vector_type(4)))  int v4i_;

union BFrag {
  v16bf v;
  unsigned int u[8];
};

// Explicit global-address-space accessors: guarantees global_load/global_store
// (LOADcnt/STOREcnt only) instead of flat_* (which also ties up DScnt/LDS path).
__device__ __forceinline__ f32x4 gload4(const float* p) {
  return *(const __attribute__((address_space(1))) f32x4*)(uintptr_t)p;
}
__device__ __forceinline__ float gload1(const float* p) {
  return *(const __attribute__((address_space(1))) float*)(uintptr_t)p;
}
__device__ __forceinline__ void gstore1(float* p, float v) {
  *(__attribute__((address_space(1))) float*)(uintptr_t)p = v;
}

// pack two f32 -> packed bf16 pair (RNE), lo in [15:0]
__device__ __forceinline__ unsigned int pk_bf16(float lo, float hi) {
  unsigned int a = __float_as_uint(lo);
  unsigned int b = __float_as_uint(hi);
  a += 0x7FFFu + ((a >> 16) & 1u);
  b += 0x7FFFu + ((b >> 16) & 1u);
  return (a >> 16) | (b & 0xFFFF0000u);
}

// TDM: DMA one 32(K) x 128(col) bf16 tile of W into LDS.
// Tensor: W rows are K-contiguous (stride 3072 elems). LDS padding: 16B per
// 64B line -> 80B line stride (bank-conflict-free ds_load_b128 afterwards).
__device__ __forceinline__ void tdm_load_tile(const unsigned short* gsrc,
                                              unsigned int lds_off) {
  unsigned long long ga = (unsigned long long)(uintptr_t)gsrc;
  v4u g0;
  g0.x = 1u;                                             // count=1 (valid user D#)
  g0.y = lds_off;                                        // lds_addr (bytes)
  g0.z = (unsigned int)ga;                               // global_addr[31:0]
  g0.w = (unsigned int)((ga >> 32) & 0x1FFFFFFu)         // global_addr[56:32]
         | (2u << 30);                                   // type = 2 ("image")
  v8i_ g1;
  g1[0] = (int)((1u << 16)      // data_size = 2 bytes
              | (1u << 20)      // pad_enable
              | (3u << 22)      // pad_interval: every 16 DWORDs (64B)
              | (3u << 25));    // pad_amount: 4 DWORDs (16B)
  g1[1] = (int)(3072u << 16);   // tensor_dim0 = 3072 (bits[79:48], low half)
  g1[2] = (int)(1024u << 16);   // tensor_dim0 hi = 0; tensor_dim1 = 1024
  g1[3] = (int)(32u << 16);     // tensor_dim1 hi = 0; tile_dim0 = 32
  g1[4] = 128;                  // tile_dim1 = 128 (tile_dim2 = 0 -> 2D)
  g1[5] = 3072;                 // tensor_dim0_stride = 3072 elems
  g1[6] = 0;
  g1[7] = 0;
  v4i_ z4 = {0, 0, 0, 0};
  v8i_ z8 = {0, 0, 0, 0, 0, 0, 0, 0};
  __builtin_amdgcn_tensor_load_to_lds(g0, g1, z4, z4, z8, 0);
}

// --- Kernel 1: W_mat f32 -> bf16 workspace, plus f32 identity row (vec(eye)) ---
__global__ void __launch_bounds__(256) prep_kernel(const float* __restrict__ W,
                                                   unsigned short* __restrict__ Wb,
                                                   float* __restrict__ ident) {
  const int idx = blockIdx.x * blockDim.x + threadIdx.x;
  if (idx < NCOLS * 3 * KSEG) {
    unsigned int u = __float_as_uint(W[idx]);
    u += 0x7FFFu + ((u >> 16) & 1u);
    Wb[idx] = (unsigned short)(u >> 16);
  }
  if (idx < KSEG) ident[idx] = ((idx >> 5) == (idx & 31)) ? 1.0f : 0.0f;
}

// --- Kernel 2: h_mat = tanh( sum_s A_shift(s) @ W_seg(s)^T + b ) via bf16 WMMA ---
// Block = 4 waves = 64 rows x 128 cols. B tiles staged in LDS by the TDM
// (double-buffered); each wave computes a 16x128 slice.
__global__ void __launch_bounds__(128) mat_gemm_kernel(
    const float* __restrict__ x_mat,          // [MROWS][1024] f32
    const unsigned short* __restrict__ Wb,    // [1024][3072] bf16 bits (row-major)
    const float* __restrict__ ident,          // [1024] f32 vec(eye)
    const float* __restrict__ b_mat,          // [1024] f32
    float* __restrict__ out_mat) {            // [MROWS][1024] f32
  __shared__ __attribute__((aligned(128))) unsigned char smB[2][BUFBYTES];

  const int lane = threadIdx.x & 31;
  const int wave = threadIdx.x >> 5;
  const int rowg = blockIdx.x >> 3;           // 512 row groups of 64
  const int colg = blockIdx.x & 7;            // 8 col groups of 128
  const int row0 = rowg * 64 + wave * 16;
  const int col0 = colg * 128;
  const int lr   = lane & 15;                 // A: row-in-tile, B: col-in-tile
  const int hi   = lane >> 4;                 // K-half select
  const int alo  = hi * 8;                    // A K offset within 32-K window

  const unsigned int lds0 = (unsigned int)(uintptr_t)&smB[0][0];
  const unsigned int lds1 = (unsigned int)(uintptr_t)&smB[1][0];

  // Per-lane A row pointers for the three shift segments (identity at edges).
  const int m  = row0 + lr;
  const int nb = m & (NPB - 1);
  const float* ApSeg[3];
  ApSeg[0] = (nb == 0)       ? ident : (x_mat + (size_t)(m - 1) * KSEG);
  ApSeg[1] = x_mat + (size_t)m * KSEG;
  ApSeg[2] = (nb == NPB - 1) ? ident : (x_mat + (size_t)(m + 1) * KSEG);

  const unsigned short* Wtile = Wb + (size_t)col0 * (3 * KSEG);

  v8f acc[8] = {};

  // Prologue: DMA first B tile; prefetch first A fragment.
  if (threadIdx.x < 32u) tdm_load_tile(Wtile, lds0);
  const float* Ap0 = ApSeg[0] + alo;
  f32x4 an0 = gload4(Ap0);
  f32x4 an1 = gload4(Ap0 + 4);
  f32x4 an2 = gload4(Ap0 + 16);
  f32x4 an3 = gload4(Ap0 + 20);

  for (int ks = 0; ks < KSTEPS; ++ks) {
    const int cur = ks & 1;
    // B tile for this step is ready once TENSORcnt drains; publish to block.
    if (threadIdx.x < 32u) __builtin_amdgcn_s_wait_tensorcnt(0);
    __syncthreads();
    // Kick off next tile's DMA into the other buffer (its last readers
    // finished before the barrier above).
    if ((threadIdx.x < 32u) && (ks + 1 < KSTEPS)) {
      const int kn = ks + 1;
      tdm_load_tile(Wtile + ((kn >> 5) * KSEG + ((kn & 31) << 5)),
                    (kn & 1) ? lds1 : lds0);
    }

    // Current A fragment from the prefetch; issue next step's A loads now.
    f32x4 a0 = an0, a1 = an1, a2 = an2, a3 = an3;
    if (ks + 1 < KSTEPS) {
      const int kn = ks + 1;
      const float* Apn = ApSeg[kn >> 5] + ((kn & 31) << 5) + alo;
      an0 = gload4(Apn);
      an1 = gload4(Apn + 4);
      an2 = gload4(Apn + 16);
      an3 = gload4(Apn + 20);
    }
    BFrag af;
    af.u[0] = pk_bf16(a0.x, a0.y); af.u[1] = pk_bf16(a0.z, a0.w);
    af.u[2] = pk_bf16(a1.x, a1.y); af.u[3] = pk_bf16(a1.z, a1.w);
    af.u[4] = pk_bf16(a2.x, a2.y); af.u[5] = pk_bf16(a2.z, a2.w);
    af.u[6] = pk_bf16(a3.x, a3.y); af.u[7] = pk_bf16(a3.z, a3.w);

    // B fragments from LDS (32B per lane, 16B-aligned, conflict-free stride).
    const unsigned char* bbase = &smB[cur][0] + hi * 16;
    BFrag bf[8];
#pragma unroll
    for (int j = 0; j < 8; ++j) {
      const uint4* p = (const uint4*)(bbase + (j * 16 + lr) * LDS_LINE);
      *(uint4*)&bf[j].u[0] = p[0];
      *(uint4*)&bf[j].u[4] = p[2];
    }
#pragma unroll
    for (int j = 0; j < 8; ++j) {
      acc[j] = __builtin_amdgcn_wmma_f32_16x16x32_bf16(
          false, af.v, false, bf[j].v, (short)0, acc[j], false, false);
    }
  }

  // Epilogue: bias + tanh + store. C layout: VGPR i -> M = i + hi*8, N = lane&15.
#pragma unroll
  for (int j = 0; j < 8; ++j) {
    const int ocol = col0 + j * 16 + lr;
    const float bias = gload1(b_mat + ocol);
#pragma unroll
    for (int i = 0; i < 8; ++i) {
      const int orow = row0 + hi * 8 + i;
      gstore1(out_mat + (size_t)orow * NCOLS + ocol, tanhf(acc[j][i] + bias));
    }
  }
}

// --- Kernel 3: h_vec path, one wave per row, full f32 ---
__global__ void __launch_bounds__(256) vec_kernel(
    const float* __restrict__ x_vec,   // [MROWS][32]
    const float* __restrict__ x_mat,   // [MROWS][1024]
    const float* __restrict__ W_vec,   // [32][64]
    const float* __restrict__ b_vec,   // [32]
    float* __restrict__ out_vec) {     // [MROWS][32]
  const int lane = threadIdx.x & 31;
  const int wave = threadIdx.x >> 5;
  const int row  = blockIdx.x * 8 + wave;
  if (row >= MROWS) return;
  const int nb = row & (NPB - 1);

  const float xv = gload1(x_vec + (size_t)row * DD + lane);

  float mvp, mvn;
  if (nb == 0) {
    mvp = xv;                                    // identity @ x_vec
  } else {
    const float* Xp = x_mat + (size_t)(row - 1) * KSEG + lane * DD;
    float s = 0.f;
#pragma unroll
    for (int j = 0; j < DD; ++j) s = fmaf(gload1(Xp + j), __shfl(xv, j, 32), s);
    mvp = s;
  }
  if (nb == NPB - 1) {
    mvn = xv;
  } else {
    const float* Xn = x_mat + (size_t)(row + 1) * KSEG + lane * DD;
    float s = 0.f;
#pragma unroll
    for (int j = 0; j < DD; ++j) s = fmaf(gload1(Xn + j), __shfl(xv, j, 32), s);
    mvn = s;
  }

  float o = gload1(b_vec + lane);
  const float* Wr = W_vec + lane * 2 * DD;
#pragma unroll
  for (int j = 0; j < DD; ++j) {
    o = fmaf(__shfl(mvp, j, 32), gload1(Wr + j), o);
    o = fmaf(__shfl(mvn, j, 32), gload1(Wr + DD + j), o);
  }
  gstore1(out_vec + (size_t)row * DD + lane, tanhf(o));
}

extern "C" void kernel_launch(void* const* d_in, const int* in_sizes, int n_in,
                              void* d_out, int out_size, void* d_ws, size_t ws_size,
                              hipStream_t stream) {
  (void)in_sizes; (void)n_in; (void)out_size; (void)ws_size;
  const float* x_vec = (const float*)d_in[0];
  const float* x_mat = (const float*)d_in[1];
  const float* W_vec = (const float*)d_in[2];
  const float* b_vec = (const float*)d_in[3];
  const float* W_mat = (const float*)d_in[4];
  const float* b_mat = (const float*)d_in[5];

  float* out_vec = (float*)d_out;              // h_vec first (tuple order)
  float* out_mat = out_vec + HVEC_ELEMS;       // then h_mat

  // Workspace layout: [0, 6291456) bf16 W_mat; then 4KB identity row (f32).
  unsigned short* Wb = (unsigned short*)d_ws;
  float* ident = (float*)((char*)d_ws + (size_t)NCOLS * 3 * KSEG * sizeof(unsigned short));

  prep_kernel<<<(NCOLS * 3 * KSEG + 255) / 256, 256, 0, stream>>>(W_mat, Wb, ident);
  mat_gemm_kernel<<<(MROWS / 64) * (NCOLS / 128), 128, 0, stream>>>(
      x_mat, Wb, ident, b_mat, out_mat);
  vec_kernel<<<MROWS / 8, 256, 0, stream>>>(x_vec, x_mat, W_vec, b_vec, out_vec);
}
